// MultiHeadAttention_35476429866000
// MI455X (gfx1250) — matrix operand order repaired and checked by reading, not compiled, not run on hardware
//
#include <hip/hip_runtime.h>
#include <hip/hip_bf16.h>
#include <stdint.h>

// MI455X (gfx1250) multi-head attention, bf16 WMMA pipeline.
// ~69 GFLOP vs ~100MB traffic -> matrix-core bound -> v_wmma_f32_16x16x32_bf16.

typedef __attribute__((ext_vector_type(16))) __bf16 v16bf;
typedef __attribute__((ext_vector_type(8)))  __bf16 v8bf;
typedef __attribute__((ext_vector_type(8)))  float  v8f;
typedef int v4i_ __attribute__((vector_size(16)));  // matches async-LDS builtin

#define AS1G __attribute__((address_space(1)))
#define AS3L __attribute__((address_space(3)))

#if defined(__has_builtin)
#  if __has_builtin(__builtin_amdgcn_global_load_async_to_lds_b128)
#    define HAVE_ASYNC_LDS 1
#  endif
#endif
#ifndef HAVE_ASYNC_LDS
#  define HAVE_ASYNC_LDS 0
#endif

static __device__ __forceinline__ void wait_async0() {
#if defined(__has_builtin) && __has_builtin(__builtin_amdgcn_s_wait_asynccnt)
    __builtin_amdgcn_s_wait_asynccnt(0);
#else
    asm volatile("s_wait_asynccnt 0" ::: "memory");
#endif
}

static __device__ __forceinline__ v8f wmma_bf16(v16bf a, v16bf b, v8f c) {
    return __builtin_amdgcn_wmma_f32_16x16x32_bf16(
        /*neg_a=*/false, a, /*neg_b=*/false, b,
        /*c_mod=*/(short)0, c, /*reuse_a=*/false, /*reuse_b=*/false);
}

static __device__ __forceinline__ v16bf pack2(v8bf lo, v8bf hi8) {
    v16bf r;
#pragma unroll
    for (int j = 0; j < 8; ++j) { r[j] = lo[j]; r[j + 8] = hi8[j]; }
    return r;
}

// ---------------------------------------------------------------------------
// fp32 -> bf16 conversion (grid-stride)
// ---------------------------------------------------------------------------
__global__ void f32_to_bf16_kernel(const float* __restrict__ src,
                                   __bf16* __restrict__ dst, int n) {
    for (int i = blockIdx.x * 256 + threadIdx.x; i < n; i += gridDim.x * 256)
        dst[i] = (__bf16)src[i];
}

// ---------------------------------------------------------------------------
// GEMM: Y[m,n] = sum_k A[m,k] * W[n,k] + bias[n]   (y = x @ W^T + b)
// M = 4096, N = K = 1024. Block tile 128x64, 256 threads = 8 waves,
// wave tile 32x32 = 2x2 WMMA, K-step 64, double-buffered LDS (1 barrier/iter),
// async global->LDS streaming (ASYNCcnt) when the toolchain exposes it.
// MODE 0: bf16 out, strided per-head layout out[(b*16+h)*2048*64 + s*ss + d*ds]
// MODE 1: fp32 out, plain [M,1024] + bias.
// ---------------------------------------------------------------------------
template <int MODE>
__global__ __launch_bounds__(256) void gemm_xwT_kernel(
    const __bf16* __restrict__ A, const __bf16* __restrict__ W,
    const float* __restrict__ bias, __bf16* __restrict__ outB,
    float* __restrict__ outF, long ss, long ds) {
    const int K = 1024, KS = 64;
    const int LDT = 72;  // padded row (bf16 elems) vs 64: kills bank conflicts
    __shared__ __align__(32) __bf16 As[2][128 * LDT];
    __shared__ __align__(32) __bf16 Bs[2][64 * LDT];

    const int t = threadIdx.x;
    const int wave = t >> 5, lane = t & 31;
    const int wm = wave >> 1, wn = wave & 1;  // 4x2 wave grid
    const int ln = lane & 15, hi = lane >> 4;
    const int rowBase = blockIdx.y * 128;
    const int colBase = blockIdx.x * 64;

    // cooperative-load mapping (per 128x64 A slab / 64x64 B slab)
    const int arA = t >> 1, acA = (t & 1) * 32;  // 32 elems (64B) per thread
    const int arB = t >> 2, acB = (t & 3) * 16;  // 16 elems (32B) per thread

    v8f acc[2][2] = {};

    auto compute = [&](int buf) {
        v16bf af[2][2], bfv[2][2];
#pragma unroll
        for (int tm = 0; tm < 2; ++tm) {
            const __bf16* p = &As[buf][(wm * 32 + tm * 16 + ln) * LDT];
#pragma unroll
            for (int kc = 0; kc < 2; ++kc)
                af[tm][kc] = pack2(*(const v8bf*)(p + kc * 32 + hi * 16),
                                   *(const v8bf*)(p + kc * 32 + hi * 16 + 8));
        }
#pragma unroll
        for (int tn = 0; tn < 2; ++tn) {
            const __bf16* p = &Bs[buf][(wn * 32 + tn * 16 + ln) * LDT];
#pragma unroll
            for (int kc = 0; kc < 2; ++kc) {
                const __bf16* q = p + kc * 32 + hi * 16;
                bfv[tn][kc] = pack2(*(const v8bf*)q, *(const v8bf*)(q + 8));
            }
        }
#pragma unroll
        for (int kc = 0; kc < 2; ++kc)
#pragma unroll
            for (int tm = 0; tm < 2; ++tm)
#pragma unroll
                for (int tn = 0; tn < 2; ++tn)
                    acc[tm][tn] = wmma_bf16(af[tm][kc], bfv[tn][kc], acc[tm][tn]);
    };

#if HAVE_ASYNC_LDS
    auto issueTile = [&](int kk, int buf) {
        const __bf16* ag = A + (long)(rowBase + arA) * K + kk + acA;
        AS1G v4i_* agp = (AS1G v4i_*)(uintptr_t)(const void*)ag;
        AS3L v4i_* alp =
            (AS3L v4i_*)(unsigned int)(uintptr_t)(void*)&As[buf][arA * LDT + acA];
        __builtin_amdgcn_global_load_async_to_lds_b128(agp, alp, 0, 0);
        __builtin_amdgcn_global_load_async_to_lds_b128(agp, alp, 16, 0);
        __builtin_amdgcn_global_load_async_to_lds_b128(agp, alp, 32, 0);
        __builtin_amdgcn_global_load_async_to_lds_b128(agp, alp, 48, 0);
        const __bf16* bg = W + (long)(colBase + arB) * K + kk + acB;
        AS1G v4i_* bgp = (AS1G v4i_*)(uintptr_t)(const void*)bg;
        AS3L v4i_* blp =
            (AS3L v4i_*)(unsigned int)(uintptr_t)(void*)&Bs[buf][arB * LDT + acB];
        __builtin_amdgcn_global_load_async_to_lds_b128(bgp, blp, 0, 0);
        __builtin_amdgcn_global_load_async_to_lds_b128(bgp, blp, 16, 0);
    };
    issueTile(0, 0);
    wait_async0();
    __syncthreads();
    int buf = 0;
    for (int kk = 0; kk < K; kk += KS) {
        if (kk + KS < K) issueTile(kk + KS, buf ^ 1);  // stream next slab
        compute(buf);                                  // WMMAs hide the DMA
        wait_async0();
        __syncthreads();
        buf ^= 1;
    }
#else
    v8bf ra[4], rb[2];
    auto gload = [&](int kk) {
        const __bf16* ap = A + (long)(rowBase + arA) * K + kk + acA;
#pragma unroll
        for (int j = 0; j < 4; ++j) ra[j] = *(const v8bf*)(ap + 8 * j);
        const __bf16* bp = W + (long)(colBase + arB) * K + kk + acB;
#pragma unroll
        for (int j = 0; j < 2; ++j) rb[j] = *(const v8bf*)(bp + 8 * j);
        __builtin_prefetch((const void*)(ap + KS), 0, 0);
        __builtin_prefetch((const void*)(bp + KS), 0, 0);
    };
    auto lstore = [&](int buf) {
#pragma unroll
        for (int j = 0; j < 4; ++j)
            *(v8bf*)&As[buf][arA * LDT + acA + 8 * j] = ra[j];
#pragma unroll
        for (int j = 0; j < 2; ++j)
            *(v8bf*)&Bs[buf][arB * LDT + acB + 8 * j] = rb[j];
    };
    gload(0);
    lstore(0);
    __syncthreads();
    int buf = 0;
    for (int kk = 0; kk < K; kk += KS) {
        if (kk + KS < K) gload(kk + KS);
        compute(buf);
        if (kk + KS < K) lstore(buf ^ 1);
        __syncthreads();
        buf ^= 1;
    }
#endif

    // epilogue: C layout -> row = i + 8*hi, col = ln (per 16x16 tile)
#pragma unroll
    for (int tm = 0; tm < 2; ++tm) {
#pragma unroll
        for (int tn = 0; tn < 2; ++tn) {
            const int n = colBase + wn * 32 + tn * 16 + ln;
            const float bv = bias[n];
#pragma unroll
            for (int i = 0; i < 8; ++i) {
                const int m = rowBase + wm * 32 + tm * 16 + i + 8 * hi;
                const float v = acc[tm][tn][i] + bv;
                if (MODE == 0) {
                    const int b = m >> 11, s = m & 2047;  // S = 2048
                    const int h = n >> 6, d = n & 63;     // hd = 64
                    const long base = (long)(b * 16 + h) * (2048L * 64L);
                    outB[base + (long)s * ss + (long)d * ds] = (__bf16)v;
                } else {
                    outF[(long)m * 1024 + n] = v;
                }
            }
        }
    }
}

// ---------------------------------------------------------------------------
// Flash-style causal attention, 64 keys/iteration. One wave per 16-query
// tile (blockDim=32: __syncthreads() is a HW NOP that still orders LDS).
// Row-sums come from an extra WMMA against an all-ones B fragment (C layout
// replicates the per-row sum across lanes), replacing 32 bpermutes/iter.
// Q,K: [B,H,S,64] bf16;  Vt: [B,H,64,S] bf16;  O: [B,S,1024] bf16.
// No 1/sqrt(d) scaling (faithful to reference).
// ---------------------------------------------------------------------------
__global__ __launch_bounds__(32) void attn_kernel(
    const __bf16* __restrict__ Q, const __bf16* __restrict__ Kk,
    const __bf16* __restrict__ Vt, __bf16* __restrict__ O) {
    const int S = 2048, HD = 64, LDP = 72;
    __shared__ __align__(32) __bf16 Plds[16 * LDP];

    const int lane = threadIdx.x & 31, ln = lane & 15, hi = lane >> 4;
    const int gid = blockIdx.x;
    const int qt = gid & 127, h = (gid >> 7) & 15, b = gid >> 11;
    const int q0 = qt * 16;

    const __bf16* qb = Q + (long)(b * 16 + h) * S * HD;
    const __bf16* kb = Kk + (long)(b * 16 + h) * S * HD;
    const __bf16* vb = Vt + (long)(b * 16 + h) * HD * S;

    // Q A-fragments for the two 32-wide K chunks of hd=64
    v16bf qa[2];
#pragma unroll
    for (int kc = 0; kc < 2; ++kc) {
        const __bf16* p = qb + (long)(q0 + ln) * HD + kc * 32;
        qa[kc] = pack2(*(const v8bf*)(p + hi * 16),
                       *(const v8bf*)(p + hi * 16 + 8));
    }
    v16bf ones;
#pragma unroll
    for (int j = 0; j < 16; ++j) ones[j] = (__bf16)1.0f;

    float m_[8], l_[8];
    v8f o[4] = {};
#pragma unroll
    for (int i = 0; i < 8; ++i) { m_[i] = -1e30f; l_[i] = 0.f; }

    const int ktLast = (q0 + 15) >> 6;  // 64 keys per iteration
    for (int kt = 0; kt <= ktLast; ++kt) {
        const int n0 = kt * 64;

        // ---- scores: four 16x16 C tiles (key quadrants), K = 64
        v8f c[4] = {};
#pragma unroll
        for (int qd = 0; qd < 4; ++qd) {
            const __bf16* kr = kb + (long)(n0 + qd * 16 + ln) * HD;
            c[qd] = wmma_bf16(qa[0], *(const v16bf*)(kr + hi * 16), c[qd]);
            c[qd] = wmma_bf16(qa[1], *(const v16bf*)(kr + 32 + hi * 16), c[qd]);
        }

        // ---- causal mask + online max (row = i + 8*hi, col = lane%16)
        float sc_[8];
#pragma unroll
        for (int i = 0; i < 8; ++i) {
            const int r = q0 + i + 8 * hi;
            float s0 = (n0 + ln > r)      ? -1e4f : c[0][i];
            float s1 = (n0 + 16 + ln > r) ? -1e4f : c[1][i];
            float s2 = (n0 + 32 + ln > r) ? -1e4f : c[2][i];
            float s3 = (n0 + 48 + ln > r) ? -1e4f : c[3][i];
            float tmax = fmaxf(fmaxf(s0, s1), fmaxf(s2, s3));
            tmax = fmaxf(tmax, __shfl_xor(tmax, 1, 32));
            tmax = fmaxf(tmax, __shfl_xor(tmax, 2, 32));
            tmax = fmaxf(tmax, __shfl_xor(tmax, 4, 32));
            tmax = fmaxf(tmax, __shfl_xor(tmax, 8, 32));
            const float mn = fmaxf(m_[i], tmax);
            sc_[i] = __expf(m_[i] - mn);
            m_[i] = mn;
            const int pr = (i + 8 * hi) * LDP + ln;
            Plds[pr]      = (__bf16)__expf(s0 - mn);
            Plds[pr + 16] = (__bf16)__expf(s1 - mn);
            Plds[pr + 32] = (__bf16)__expf(s2 - mn);
            Plds[pr + 48] = (__bf16)__expf(s3 - mn);
        }
#pragma unroll
        for (int t = 0; t < 4; ++t)
#pragma unroll
            for (int i = 0; i < 8; ++i) o[t][i] *= sc_[i];

        __syncthreads();  // single-wave WG: HW NOP, orders ds stores/loads

        // ---- reload P as two 16x32 A fragments
        const __bf16* pr = &Plds[ln * LDP];
        v16bf pf[2];
        pf[0] = pack2(*(const v8bf*)(pr + hi * 16),
                      *(const v8bf*)(pr + hi * 16 + 8));
        pf[1] = pack2(*(const v8bf*)(pr + 32 + hi * 16),
                      *(const v8bf*)(pr + 32 + hi * 16 + 8));

        // ---- row-sum of P via WMMA against ones (replaces 32 bpermutes)
        v8f rsum = {};
        rsum = wmma_bf16(pf[0], ones, rsum);
        rsum = wmma_bf16(pf[1], ones, rsum);
#pragma unroll
        for (int i = 0; i < 8; ++i) l_[i] = l_[i] * sc_[i] + rsum[i];

        // ---- O += P(16x64) @ V(64x64); Vt pre-transposed -> contig B frags
#pragma unroll
        for (int t = 0; t < 4; ++t) {
            const __bf16* vr = vb + (long)(t * 16 + ln) * S + n0;
            o[t] = wmma_bf16(pf[0], *(const v16bf*)(vr + hi * 16), o[t]);
            o[t] = wmma_bf16(pf[1], *(const v16bf*)(vr + 32 + hi * 16), o[t]);
        }
        __syncthreads();
    }

    // ---- normalize and store O in [B,S,1024] bf16
#pragma unroll
    for (int t = 0; t < 4; ++t) {
#pragma unroll
        for (int i = 0; i < 8; ++i) {
            const float v = o[t][i] / l_[i];
            const int s = q0 + i + 8 * hi;
            const int d = h * 64 + t * 16 + ln;
            O[(long)(b * 2048 + s) * 1024 + d] = (__bf16)v;
        }
    }
}

// ---------------------------------------------------------------------------
// launch
// ---------------------------------------------------------------------------
extern "C" void kernel_launch(void* const* d_in, const int* in_sizes, int n_in,
                              void* d_out, int out_size, void* d_ws, size_t ws_size,
                              hipStream_t stream) {
    (void)in_sizes; (void)n_in; (void)out_size; (void)ws_size;
    const float* x  = (const float*)d_in[0];
    // d_in[1] = mask (causal, applied analytically in-kernel)
    const float* Wq = (const float*)d_in[2];
    const float* bq = (const float*)d_in[3];
    const float* Wk = (const float*)d_in[4];
    const float* bk = (const float*)d_in[5];
    const float* Wv = (const float*)d_in[6];
    const float* bv = (const float*)d_in[7];
    const float* Wo = (const float*)d_in[8];
    const float* bo = (const float*)d_in[9];

    char* ws = (char*)d_ws;
    const size_t MB = (size_t)1 << 20;
    __bf16* xb  = (__bf16*)(ws);            //  8 MB  x bf16 [4096,1024]
    __bf16* wqb = (__bf16*)(ws +  8 * MB);  //  2 MB
    __bf16* wkb = (__bf16*)(ws + 10 * MB);  //  2 MB
    __bf16* wvb = (__bf16*)(ws + 12 * MB);  //  2 MB
    __bf16* wob = (__bf16*)(ws + 14 * MB);  //  2 MB
    __bf16* Qb  = (__bf16*)(ws + 16 * MB);  //  8 MB  [B,H,S,64]
    __bf16* Kb  = (__bf16*)(ws + 24 * MB);  //  8 MB  [B,H,S,64]
    __bf16* Vt  = (__bf16*)(ws + 32 * MB);  //  8 MB  [B,H,64,S]
    __bf16* Ob  = (__bf16*)(ws + 40 * MB);  //  8 MB  [B,S,1024]

    f32_to_bf16_kernel<<<2048, 256, 0, stream>>>(x,  xb,  4096 * 1024);
    f32_to_bf16_kernel<<<1024, 256, 0, stream>>>(Wq, wqb, 1024 * 1024);
    f32_to_bf16_kernel<<<1024, 256, 0, stream>>>(Wk, wkb, 1024 * 1024);
    f32_to_bf16_kernel<<<1024, 256, 0, stream>>>(Wv, wvb, 1024 * 1024);
    f32_to_bf16_kernel<<<1024, 256, 0, stream>>>(Wo, wob, 1024 * 1024);

    dim3 gg(1024 / 64, 4096 / 128);  // (N tiles, M tiles)
    gemm_xwT_kernel<0><<<gg, 256, 0, stream>>>(xb, wqb, bq, Qb, nullptr, 64, 1);
    gemm_xwT_kernel<0><<<gg, 256, 0, stream>>>(xb, wkb, bk, Kb, nullptr, 64, 1);
    gemm_xwT_kernel<0><<<gg, 256, 0, stream>>>(xb, wvb, bv, Vt, nullptr, 1, 2048);

    attn_kernel<<<2 * 16 * 128, 32, 0, stream>>>(Qb, Kb, Vt, Ob);

    gemm_xwT_kernel<1><<<gg, 256, 0, stream>>>(Ob, wob, bo, nullptr,
                                               (float*)d_out, 0, 0);
}
